// GRURNN_81097572483831
// MI455X (gfx1250) — matrix operand order repaired and checked by reading, not compiled, not run on hardware
//
#include <hip/hip_runtime.h>
#include <hip/hip_bf16.h>
#include <hip/hip_fp16.h>

// ---------------- problem constants ----------------
#define BB   256
#define TT   2048
#define II   192
#define H0   128
#define H1   64
#define OO   2

// WMMA tiling
#define KT_I   6      // 192/32
#define KT_H0  4      // 128/32
#define KT_H1  2      // 64/32
#define JT0    8      // H0/16
#define JT1    4      // H1/16

// packed-weight ws layout (in halves)
#define OFF_WIH0  0
#define SZ_WIH0   (24*6*512)
#define OFF_WHH0  (OFF_WIH0 + SZ_WIH0)
#define SZ_WHH0   (24*4*512)
#define OFF_WIH1  (OFF_WHH0 + SZ_WHH0)
#define SZ_WIH1   (12*4*512)
#define OFF_WHH1  (OFF_WIH1 + SZ_WIH1)
#define SZ_WHH1   (12*2*512)
#define N_TILES_TOTAL (144 + 96 + 48 + 24)   // 312

// LDS strides: rows 16B-aligned so A-fragments load as ds b128;
// 136 halves = 272B (68 dw), 72 halves = 144B (36 dw): 4*lane bank rotation
// keeps quarter-wave b128 loads conflict-free.
#define H0S 136   // halves per row of sh0
#define H1S 72    // halves per row of sh1
#define HNS 133   // floats per row of pre-LN scratch (odd dw stride for LN walks)

typedef __attribute__((ext_vector_type(16))) _Float16 v16h;
typedef __attribute__((ext_vector_type(8)))  float    v8f;
typedef __attribute__((ext_vector_type(4)))  float    v4f;

union Frag16 { v16h v; v4f q[2]; };

#define WMMA_F16(acc, a, b) \
    acc = __builtin_amdgcn_wmma_f32_16x16x32_f16(false, (a), false, (b), (short)0, (acc), false, false)

// --------------------------------------------------------------------------
// Kernel 1: pack weights into WMMA B-fragment layout (f16).
// B tile K(32) x N(16): lane l -> column N = l&15, halves i -> K = (l>>4)*8 + (i&7) + (i>>3)*16.
// B[k][n] = W[nBase+n][kBase+k]  (C = A * W^T).
// --------------------------------------------------------------------------
__global__ __launch_bounds__(512)
void gru_pack_weights(const float* __restrict__ Wih0,
                      const float* __restrict__ Whh0,
                      const float* __restrict__ Wih1,
                      const float* __restrict__ Whh1,
                      _Float16* __restrict__ wsh) {
    int tile = blockIdx.x;
    int lane = threadIdx.x >> 4;   // 0..31
    int i    = threadIdx.x & 15;   // 0..15

    const float* W; int ncols; long base; int nt, kt;
    if (tile < 144)      { W = Wih0; ncols = II;  int t = tile;       nt = t/6; kt = t%6; base = OFF_WIH0 + (long)t*512; }
    else if (tile < 240) { W = Whh0; ncols = H0;  int t = tile - 144; nt = t/4; kt = t%4; base = OFF_WHH0 + (long)t*512; }
    else if (tile < 288) { W = Wih1; ncols = H0;  int t = tile - 240; nt = t/4; kt = t%4; base = OFF_WIH1 + (long)t*512; }
    else                 { W = Whh1; ncols = H1;  int t = tile - 288; nt = t/2; kt = t%2; base = OFF_WHH1 + (long)t*512; }

    int n = nt*16 + (lane & 15);
    int k = kt*32 + (lane >> 4)*8 + (i & 7) + (i >> 3)*16;
    wsh[base + lane*16 + i] = (_Float16)W[(long)n * ncols + k];
}

// --------------------------------------------------------------------------
// A fragment (16x32 f16): lane l -> row M = l&15,
// halves 0..7 -> K = (l>>4)*8 + i, halves 8..15 -> K = 16 + (l>>4)*8 + (i-8).
// --------------------------------------------------------------------------
__device__ __forceinline__ v16h load_a_global_x(const float* __restrict__ p_row0,
                                                long rowStride, int lane, int kBase) {
    const int row = lane & 15;
    const int klo = (lane >> 4) * 8;
    const float* p = p_row0 + (long)row * rowStride + kBase + klo;
    const v4f* q0 = (const v4f*)p;          // 32B aligned
    const v4f* q1 = (const v4f*)(p + 16);
    v4f a0 = __builtin_nontemporal_load(q0);
    v4f a1 = __builtin_nontemporal_load(q0 + 1);
    v4f a2 = __builtin_nontemporal_load(q1);
    v4f a3 = __builtin_nontemporal_load(q1 + 1);
    v16h a;
#pragma unroll
    for (int i = 0; i < 4; ++i) a[i]      = (_Float16)a0[i];
#pragma unroll
    for (int i = 0; i < 4; ++i) a[4 + i]  = (_Float16)a1[i];
#pragma unroll
    for (int i = 0; i < 4; ++i) a[8 + i]  = (_Float16)a2[i];
#pragma unroll
    for (int i = 0; i < 4; ++i) a[12 + i] = (_Float16)a3[i];
    return a;
}

__device__ __forceinline__ v16h load_a_lds_f16(const _Float16* __restrict__ h,
                                               int rowStride, int mBase, int lane, int kBase) {
    const int row = mBase + (lane & 15);
    const int klo = (lane >> 4) * 8;
    const _Float16* p = h + (long)row * rowStride + kBase + klo;   // 16B aligned
    Frag16 u;
    u.q[0] = *(const v4f*)p;          // halves 0..7
    u.q[1] = *(const v4f*)(p + 16);   // halves 8..15
    return u.v;
}

__device__ __forceinline__ v16h load_b_packed(const _Float16* __restrict__ tb, int lane) {
    Frag16 u;
    const v4f* p = (const v4f*)(tb + lane * 16);
    u.q[0] = p[0];
    u.q[1] = p[1];
    return u.v;
}

__device__ __forceinline__ float sigmoidf_(float x) { return 1.0f / (1.0f + __expf(-x)); }

// --------------------------------------------------------------------------
// Kernel 2: persistent recurrent kernel. One workgroup, 16 waves (512 thr),
// hidden states resident in LDS for all 2048 steps. Each wave owns one fixed
// m-tile (mt = wave) so A fragments are converted once per step and reused
// across all j-tiles. x_{t+1} is prefetched (global_prefetch) while the
// current step's WMMAs run.
// --------------------------------------------------------------------------
__global__ __launch_bounds__(512)
void gru_recurrent(const float* __restrict__ x,
                   const float* __restrict__ b_ih0, const float* __restrict__ b_hh0,
                   const float* __restrict__ b_ih1, const float* __restrict__ b_hh1,
                   const float* __restrict__ g0, const float* __restrict__ be0,
                   const float* __restrict__ g1, const float* __restrict__ be1,
                   const float* __restrict__ fcW, const float* __restrict__ fcb,
                   const _Float16* __restrict__ wsh,
                   float* __restrict__ out) {
    __shared__ _Float16 sh0[BB * H0S];   // post-LN h0, f16  (69,632 B)
    __shared__ _Float16 sh1[BB * H1S];   // post-LN h1, f16  (36,864 B)
    __shared__ float    shn[BB * HNS];   // pre-LN scratch, f32 (136,192 B), reused by both layers

    const int tid  = threadIdx.x;
    const int lane = tid & 31;
    const int wave = tid >> 5;           // 0..15 == this wave's m-tile
    const int mBase = wave * 16;

    const _Float16* Wih0B = wsh + OFF_WIH0;
    const _Float16* Whh0B = wsh + OFF_WHH0;
    const _Float16* Wih1B = wsh + OFF_WIH1;
    const _Float16* Whh1B = wsh + OFF_WHH1;

    for (int idx = tid; idx < BB * H0S; idx += 512) sh0[idx] = (_Float16)0.0f;
    for (int idx = tid; idx < BB * H1S; idx += 512) sh1[idx] = (_Float16)0.0f;
    __syncthreads();

    for (int t = 0; t < TT; ++t) {
        // ---------------- Phase A: layer-0 GEMMs + gates ----------------
        {
            // hoist A fragments: x_t (K=192 -> 6 frags), h0 (K=128 -> 4 frags)
            const float* xrow0 = x + (long)mBase * (TT * II) + (long)t * II;
            v16h ax[KT_I];
#pragma unroll
            for (int kt = 0; kt < KT_I; ++kt)
                ax[kt] = load_a_global_x(xrow0, (long)TT * II, lane, kt * 32);
            v16h ah[KT_H0];
#pragma unroll
            for (int kt = 0; kt < KT_H0; ++kt)
                ah[kt] = load_a_lds_f16(sh0, H0S, mBase, lane, kt * 32);

            // prefetch this wave's 16 rows of x_{t+1} (16 rows x 768B,
            // 128B granules: 96 addresses over 32 lanes = 3 prefetches)
            if (t + 1 < TT) {
                const float* xn = xrow0 + II;   // same rows, next timestep
#pragma unroll
                for (int pp = 0; pp < 3; ++pp) {
                    int idx = pp * 32 + lane;   // 0..95
                    int row = idx / 6;
                    int seg = idx % 6;
                    __builtin_prefetch(xn + (long)row * (TT * II) + seg * 32, 0, 3);
                }
            }

#pragma unroll 1
            for (int j = 0; j < JT0; ++j) {
                v8f ar = {}, az = {}, agin = {}, aghn = {};
#pragma unroll
                for (int kt = 0; kt < KT_I; ++kt) {
                    v16h br = load_b_packed(Wih0B + ((long)((j     ) * KT_I + kt)) * 512, lane);
                    v16h bz = load_b_packed(Wih0B + ((long)((j +  8) * KT_I + kt)) * 512, lane);
                    v16h bn = load_b_packed(Wih0B + ((long)((j + 16) * KT_I + kt)) * 512, lane);
                    WMMA_F16(ar, ax[kt], br);
                    WMMA_F16(az, ax[kt], bz);
                    WMMA_F16(agin, ax[kt], bn);
                }
#pragma unroll
                for (int kt = 0; kt < KT_H0; ++kt) {
                    v16h br = load_b_packed(Whh0B + ((long)((j     ) * KT_H0 + kt)) * 512, lane);
                    v16h bz = load_b_packed(Whh0B + ((long)((j +  8) * KT_H0 + kt)) * 512, lane);
                    v16h bn = load_b_packed(Whh0B + ((long)((j + 16) * KT_H0 + kt)) * 512, lane);
                    WMMA_F16(ar, ah[kt], br);
                    WMMA_F16(az, ah[kt], bz);
                    WMMA_F16(aghn, ah[kt], bn);
                }

                const int c   = j * 16 + (lane & 15);
                const float br_  = b_ih0[c] + b_hh0[c];
                const float bz_  = b_ih0[H0 + c] + b_hh0[H0 + c];
                const float bin_ = b_ih0[2 * H0 + c];
                const float bhn_ = b_hh0[2 * H0 + c];
#pragma unroll
                for (int v = 0; v < 8; ++v) {
                    const int m = mBase + v + 8 * (lane >> 4);
                    const float r = sigmoidf_(ar[v] + br_);
                    const float z = sigmoidf_(az[v] + bz_);
                    const float n = tanhf(agin[v] + bin_ + r * (aghn[v] + bhn_));
                    const float hold = (float)sh0[(long)m * H0S + c];
                    shn[(long)m * HNS + c] = (1.0f - z) * n + z * hold;
                }
            }
        }
        __syncthreads();

        // ---------------- Phase B: LayerNorm layer-0 ----------------
        // 2 threads per row (adjacent lanes), partials merged via shfl_xor.
        {
            const int r    = tid >> 1;
            const int half = tid & 1;
            const float* base = &shn[(long)r * HNS + half * 64];
            float s = 0.0f;
#pragma unroll 4
            for (int i = 0; i < 64; ++i) s += base[i];
            s += __shfl_xor(s, 1);
            const float mu = s * (1.0f / H0);
            float vv = 0.0f;
#pragma unroll 4
            for (int i = 0; i < 64; ++i) { float d = base[i] - mu; vv += d * d; }
            vv += __shfl_xor(vv, 1);
            const float inv = rsqrtf(vv * (1.0f / H0) + 1e-5f);
#pragma unroll 4
            for (int i = 0; i < 64; ++i) {
                const int c = half * 64 + i;
                sh0[(long)r * H0S + c] = (_Float16)((base[i] - mu) * inv * g0[c] + be0[c]);
            }
        }
        __syncthreads();

        // ---------------- Phase C: layer-1 GEMMs + gates ----------------
        {
            v16h ah0[KT_H0];
#pragma unroll
            for (int kt = 0; kt < KT_H0; ++kt)
                ah0[kt] = load_a_lds_f16(sh0, H0S, mBase, lane, kt * 32);
            v16h ah1[KT_H1];
#pragma unroll
            for (int kt = 0; kt < KT_H1; ++kt)
                ah1[kt] = load_a_lds_f16(sh1, H1S, mBase, lane, kt * 32);

#pragma unroll 1
            for (int j = 0; j < JT1; ++j) {
                v8f ar = {}, az = {}, agin = {}, aghn = {};
#pragma unroll
                for (int kt = 0; kt < KT_H0; ++kt) {
                    v16h br = load_b_packed(Wih1B + ((long)((j    ) * KT_H0 + kt)) * 512, lane);
                    v16h bz = load_b_packed(Wih1B + ((long)((j + 4) * KT_H0 + kt)) * 512, lane);
                    v16h bn = load_b_packed(Wih1B + ((long)((j + 8) * KT_H0 + kt)) * 512, lane);
                    WMMA_F16(ar, ah0[kt], br);
                    WMMA_F16(az, ah0[kt], bz);
                    WMMA_F16(agin, ah0[kt], bn);
                }
#pragma unroll
                for (int kt = 0; kt < KT_H1; ++kt) {
                    v16h br = load_b_packed(Whh1B + ((long)((j    ) * KT_H1 + kt)) * 512, lane);
                    v16h bz = load_b_packed(Whh1B + ((long)((j + 4) * KT_H1 + kt)) * 512, lane);
                    v16h bn = load_b_packed(Whh1B + ((long)((j + 8) * KT_H1 + kt)) * 512, lane);
                    WMMA_F16(ar, ah1[kt], br);
                    WMMA_F16(az, ah1[kt], bz);
                    WMMA_F16(aghn, ah1[kt], bn);
                }

                const int c   = j * 16 + (lane & 15);
                const float br_  = b_ih1[c] + b_hh1[c];
                const float bz_  = b_ih1[H1 + c] + b_hh1[H1 + c];
                const float bin_ = b_ih1[2 * H1 + c];
                const float bhn_ = b_hh1[2 * H1 + c];
#pragma unroll
                for (int v = 0; v < 8; ++v) {
                    const int m = mBase + v + 8 * (lane >> 4);
                    const float r = sigmoidf_(ar[v] + br_);
                    const float z = sigmoidf_(az[v] + bz_);
                    const float n = tanhf(agin[v] + bin_ + r * (aghn[v] + bhn_));
                    const float hold = (float)sh1[(long)m * H1S + c];
                    shn[(long)m * HNS + c] = (1.0f - z) * n + z * hold;
                }
            }
        }
        __syncthreads();

        // ---------------- Phase D: LayerNorm layer-1 ----------------
        {
            const int r    = tid >> 1;
            const int half = tid & 1;
            const float* base = &shn[(long)r * HNS + half * 32];
            float s = 0.0f;
#pragma unroll 4
            for (int i = 0; i < 32; ++i) s += base[i];
            s += __shfl_xor(s, 1);
            const float mu = s * (1.0f / H1);
            float vv = 0.0f;
#pragma unroll 4
            for (int i = 0; i < 32; ++i) { float d = base[i] - mu; vv += d * d; }
            vv += __shfl_xor(vv, 1);
            const float inv = rsqrtf(vv * (1.0f / H1) + 1e-5f);
#pragma unroll 4
            for (int i = 0; i < 32; ++i) {
                const int c = half * 32 + i;
                sh1[(long)r * H1S + c] = (_Float16)((base[i] - mu) * inv * g1[c] + be1[c]);
            }
        }
        __syncthreads();
    }

    // ---------------- Final FC: out = h1 @ fcW^T + fcb ----------------
    if (tid < BB * OO) {
        const int row = tid >> 1;
        const int o   = tid & 1;
        float acc = fcb[o];
#pragma unroll 4
        for (int k = 0; k < H1; ++k)
            acc += (float)sh1[(long)row * H1S + k] * fcW[o * H1 + k];
        out[row * OO + o] = acc;
    }
}

// --------------------------------------------------------------------------
extern "C" void kernel_launch(void* const* d_in, const int* in_sizes, int n_in,
                              void* d_out, int out_size, void* d_ws, size_t ws_size,
                              hipStream_t stream) {
    const float* x     = (const float*)d_in[0];
    const float* Wih0  = (const float*)d_in[1];
    const float* Whh0  = (const float*)d_in[2];
    const float* bih0  = (const float*)d_in[3];
    const float* bhh0  = (const float*)d_in[4];
    const float* Wih1  = (const float*)d_in[5];
    const float* Whh1  = (const float*)d_in[6];
    const float* bih1  = (const float*)d_in[7];
    const float* bhh1  = (const float*)d_in[8];
    const float* g0    = (const float*)d_in[9];
    const float* be0   = (const float*)d_in[10];
    const float* g1    = (const float*)d_in[11];
    const float* be1   = (const float*)d_in[12];
    const float* fcW   = (const float*)d_in[13];
    const float* fcb   = (const float*)d_in[14];

    _Float16* wsh = (_Float16*)d_ws;
    float* out = (float*)d_out;

    gru_pack_weights<<<N_TILES_TOTAL, 512, 0, stream>>>(Wih0, Whh0, Wih1, Whh1, wsh);
    gru_recurrent<<<1, 512, 0, stream>>>(x, bih0, bhh0, bih1, bhh1,
                                         g0, be0, g1, be1, fcW, fcb, wsh, out);
}
